// StructureModel_SSL_32461362823533
// MI455X (gfx1250) — matrix operand order repaired and checked by reading, not compiled, not run on hardware
//
#include <hip/hip_runtime.h>
#include <hip/hip_bf16.h>

// ---------------------------------------------------------------------------
// StructureModel_SSL on MI455X (gfx1250): fused WMMA implementation, v3.
//  - weights pre-swizzled into per-lane-contiguous B-fragment layout (v16h load)
//  - A-fragments gathered directly from row-major LDS with b64/b128 loads
//  - global->LDS staging via GLOBAL_LOAD_ASYNC_TO_LDS_B128 (ASYNCcnt path)
//  - e1 bias folded into the GEMM via a constant-1 pad slot in the K dim
//  - MHA: flash-style online softmax; V stored head-transposed for vector loads
// ---------------------------------------------------------------------------

typedef _Float16 h16;
typedef __attribute__((ext_vector_type(16))) _Float16 v16h;
typedef __attribute__((ext_vector_type(8)))  float    v8f;

#define NODES_TOT 32768
#define BGRAPH    64
#define NGRAPH    512
#define DEG       32
#define NSPLIT    4          // workgroups per graph (each owns 128 dst nodes)
#define NOWN      128

// -------------------------- small helpers ----------------------------------

__device__ __forceinline__ v8f zero8() {
  v8f z = {0.f,0.f,0.f,0.f,0.f,0.f,0.f,0.f};
  return z;
}
__device__ __forceinline__ v16h zero16() {
  v16h z = {(h16)0.f,(h16)0.f,(h16)0.f,(h16)0.f,(h16)0.f,(h16)0.f,(h16)0.f,(h16)0.f,
            (h16)0.f,(h16)0.f,(h16)0.f,(h16)0.f,(h16)0.f,(h16)0.f,(h16)0.f,(h16)0.f};
  return z;
}

__device__ __forceinline__ v8f wmma16(v16h a, v16h b, v8f c) {
  // D = A(16x32 f16) * B(32x16 f16) + C(16x16 f32)
  return __builtin_amdgcn_wmma_f32_16x16x32_f16(false, a, false, b,
                                                (short)0, c, false, false);
}

__device__ __forceinline__ float siluf(float x) {
  return x * (1.f / (1.f + __expf(-x)));
}

// Async DMA: copy 16 bytes global -> LDS without a VGPR round trip (ASYNCcnt).
__device__ __forceinline__ void asyncLds128(void* ldsDst, const void* gSrc) {
  const unsigned int loff = (unsigned int)(uintptr_t)ldsDst;   // addrspace(3) offset
  asm volatile("global_load_async_to_lds_b128 %0, %1, off"
               :: "v"(loff), "v"((unsigned long long)(uintptr_t)gSrc)
               : "memory");
}
__device__ __forceinline__ void asyncWait0() {
  asm volatile("s_wait_asynccnt 0x0" ::: "memory");
}

// A fragment (16x32 f16) for lane's row from a contiguous row pointer.
// Per ISA 7.12.2: lane L holds row L&15; element idx holds
//   k = k0 + ((idx<8)?idx:idx+8) + 8*(L>=16)  -> two 8-half contiguous runs.
__device__ __forceinline__ v16h rowAfrag(const h16* rowp, int k0, int hi) {
  union { v16h v; uint2 q[4]; } u;
  const char* p = (const char*)(rowp + k0 + 8 * hi);
  u.q[0] = *(const uint2*)(p);
  u.q[1] = *(const uint2*)(p + 8);
  u.q[2] = *(const uint2*)(p + 32);
  u.q[3] = *(const uint2*)(p + 40);
  return u.v;
}

// B fragment from pre-swizzled weight buffer: lane's 16 halves contiguous.
__device__ __forceinline__ v16h bfragSwz(const h16* Ws, int kc, int nt, int lane) {
  return *(const v16h*)(Ws + (size_t)(((kc << 2) + nt) * 32 + lane) * 16);
}

// Scatter one C-layout column (8 rows r+8*hi, col = nt*16+lrow) into an
// A-fragment-ready buffer laid out as [kc][32 lanes][16 halves].
__device__ __forceinline__ void scatterColToA(h16* mA, int nt, int lrow, int hi,
                                              const float* v8) {
  const int kcM = nt >> 1;
  const int kk  = ((nt & 1) << 4) + lrow;
  const int hiM = (kk >> 3) & 1;
  const int eM  = kk - 8 * ((kk >> 4) + hiM);
#pragma unroll
  for (int r = 0; r < 8; ++r)
    mA[(size_t)(kcM * 32 + (r + 8 * hi) + 16 * hiM) * 16 + eM] = (h16)v8[r];
}

// Generic (layer-0) A-fragment gathers:
// f = [h_src | h_dst | radial | ef | 1(bias slot) | 0]
template<int DINc>
__device__ __forceinline__ v16h gatherEdgeAfrag(const h16* srow, const h16* drow,
                                                float radial, float ef,
                                                int k0, int hi) {
  v16h a;
#pragma unroll
  for (int idx = 0; idx < 16; ++idx) {
    const int k = k0 + ((idx < 8) ? idx : idx + 8) + 8 * hi;
    h16 v;
    if      (k < DINc)          v = srow[k];
    else if (k < 2 * DINc)      v = drow[k - DINc];
    else if (k == 2 * DINc)     v = (h16)radial;
    else if (k == 2 * DINc + 1) v = (h16)ef;
    else if (k == 2 * DINc + 2) v = (h16)1.f;      // bias slot
    else                        v = (h16)0.f;
    a[idx] = v;
  }
  return a;
}
template<int DINc>
__device__ __forceinline__ v16h gatherNodeAfrag(const h16* hrow, const h16* grow,
                                                int k0, int hi) {
  v16h a;
#pragma unroll
  for (int idx = 0; idx < 16; ++idx) {
    const int k = k0 + ((idx < 8) ? idx : idx + 8) + 8 * hi;
    h16 v;
    if      (k < DINc)      v = hrow[k];
    else if (k < DINc + 64) v = grow[k - DINc];
    else                    v = (h16)0.f;
    a[idx] = v;
  }
  return a;
}

// ------------------------- weight prep kernel ------------------------------

struct PrepArgs {
  const float* node_feat; const float* coord;
  const float* e1W[6]; const float* e2W[6]; const float* c1W[6];
  const float* n1W[6]; const float* n2W[6];
  const float* e1b[6];
  const float* qW; const float* kW; const float* vW;
  h16* h0; float* x0;
  h16* e1o[6]; h16* e2o[6]; h16* c1o[6]; h16* n1o[6]; h16* n2o[6];
  h16* qo; h16* ko; h16* vo;
  float* poolO;
};

// Swizzle row-major [rows][64] f32 weights into B-fragment layout
// [(kc*4+nt)*32+lane][16] f16, zero-padded to K rows. Optionally place a
// bias vector at padded row `biasRow` (bias folding via constant-1 A slot).
__device__ void swizzleW(const float* W, const float* bias, int biasRow,
                         h16* out, int K, int rows, long tid0, long stride) {
  const long total = (long)K * 64;
  for (long o = tid0; o < total; o += stride) {
    const int idx  = (int)(o & 15);
    const int lane = (int)((o >> 4) & 31);
    const int nt   = (int)((o >> 9) & 3);
    const int kc   = (int)(o >> 11);
    const int krow = kc * 32 + ((lane >> 4) << 3) + ((idx < 8) ? idx : idx + 8);
    const int col  = nt * 16 + (lane & 15);
    float v;
    if      (krow < rows)     v = W[krow * 64 + col];
    else if (krow == biasRow) v = bias[col];
    else                      v = 0.f;
    out[o] = (h16)v;
  }
}

__global__ __launch_bounds__(256) void prep_kernel(PrepArgs a) {
  const long stride = (long)gridDim.x * blockDim.x;
  const long tid0   = (long)blockIdx.x * blockDim.x + threadIdx.x;
  for (long i = tid0; i < (long)NODES_TOT * 20; i += stride)
    a.h0[i] = (h16)a.node_feat[i];
  for (long i = tid0; i < (long)NODES_TOT * 4; i += stride) {
    const long n = i >> 2; const int c = (int)(i & 3);
    a.x0[i] = (c < 3) ? a.coord[n * 3 + c] : 0.f;
  }
  for (int l = 0; l < 6; ++l) {
    const int din = l ? 64 : 20;
    const int FK  = l ? 160 : 64;
    const int KN1 = l ? 128 : 96;
    // e1 bias folded into padded row 2*din+2 (matches constant-1 A slot)
    swizzleW(a.e1W[l], a.e1b[l], 2 * din + 2, a.e1o[l], FK, 2 * din + 2, tid0, stride);
    swizzleW(a.e2W[l], nullptr, -1, a.e2o[l], 64,  64,       tid0, stride);
    swizzleW(a.c1W[l], nullptr, -1, a.c1o[l], 64,  64,       tid0, stride);
    swizzleW(a.n1W[l], nullptr, -1, a.n1o[l], KN1, din + 64, tid0, stride);
    swizzleW(a.n2W[l], nullptr, -1, a.n2o[l], 64,  64,       tid0, stride);
  }
  swizzleW(a.qW, nullptr, -1, a.qo, 64, 64, tid0, stride);
  swizzleW(a.kW, nullptr, -1, a.ko, 64, 64, tid0, stride);
  swizzleW(a.vW, nullptr, -1, a.vo, 64, 64, tid0, stride);
  for (long i = tid0; i < 64 * 64; i += stride) a.poolO[i] = 0.f;
}

// --------------------------- EGNN layer ------------------------------------

struct LayerW {
  const h16 *e1W, *e2W, *c1W, *n1W, *n2W;                 // f16, swizzled
  const float *e2b, *c1b, *c2w, *n1b, *n2b;               // f32 (e1b folded)
};

constexpr size_t edge_lds_bytes(int DIN, int FK, int KN1) {
  return (size_t)512 * DIN * 2               // hT
       + (size_t)8192                        // xT (pad to 4 floats)
       + (size_t)NOWN * 64 * 2               // haggT
       + (size_t)(FK * 64 + 3 * 4096 + KN1 * 64) * 2   // swizzled weights
       + (size_t)384 * 4                     // bias vectors
       + (size_t)8 * (2048 + 256 + 64);      // per-wave: mA + xdT + coefT
}

template<int DIN, int FK, int KN1>
__global__ __launch_bounds__(256) void egnn_layer_kernel(
    const h16*  __restrict__ h_in,  const float* __restrict__ x_in,
    h16*        __restrict__ h_out, float*       __restrict__ x_out,
    const int*  __restrict__ edge_src, const float* __restrict__ edge_feat,
    LayerW w)
{
  constexpr int KC1 = FK / 32;
  constexpr int KC2 = KN1 / 32;

  extern __shared__ char smem[];
  h16*   hT    = (h16*)smem;
  float* xT    = (float*)(smem + (size_t)512 * DIN * 2);
  h16*   haggT = (h16*)((char*)xT + 8192);
  h16*   e1Ws  = haggT + NOWN * 64;
  h16*   e2Ws  = e1Ws + FK * 64;
  h16*   c1Ws  = e2Ws + 4096;
  h16*   n2Ws  = c1Ws + 4096;
  h16*   n1Ws  = n2Ws + 4096;
  float* biasS = (float*)(n1Ws + KN1 * 64);     // -|e2b|c1b|c2w|n1b|n2b
  char*  pws   = (char*)(biasS + 384);

  const int tid   = threadIdx.x;
  const int graph = blockIdx.x >> 2;
  const int part  = blockIdx.x & 3;
  const int g0    = graph * NGRAPH;
  const int n0    = part * NOWN;

  { // async DMA staging: global -> LDS directly (16B per lane per op)
    const uint4* hs = (const uint4*)(h_in + (size_t)g0 * DIN);
    uint4* hd = (uint4*)hT;
    for (int i = tid; i < 512 * DIN / 8; i += 256) asyncLds128(hd + i, hs + i);
    const uint4* xs = (const uint4*)(x_in + (size_t)g0 * 4);
    uint4* xd = (uint4*)xT;
    for (int i = tid; i < 512; i += 256) asyncLds128(xd + i, xs + i);
    const uint4* ws1 = (const uint4*)w.e1W; uint4* wd1 = (uint4*)e1Ws;
    for (int i = tid; i < FK * 64 / 8; i += 256) asyncLds128(wd1 + i, ws1 + i);
    const uint4* ws2 = (const uint4*)w.e2W; uint4* wd2 = (uint4*)e2Ws;
    const uint4* ws3 = (const uint4*)w.c1W; uint4* wd3 = (uint4*)c1Ws;
    const uint4* ws5 = (const uint4*)w.n2W; uint4* wd5 = (uint4*)n2Ws;
    for (int i = tid; i < 512; i += 256) {
      asyncLds128(wd2 + i, ws2 + i);
      asyncLds128(wd3 + i, ws3 + i);
      asyncLds128(wd5 + i, ws5 + i);
    }
    const uint4* ws4 = (const uint4*)w.n1W; uint4* wd4 = (uint4*)n1Ws;
    for (int i = tid; i < KN1 * 64 / 8; i += 256) asyncLds128(wd4 + i, ws4 + i);
    if (tid < 64) {
      biasS[64 + tid]  = w.e2b[tid]; biasS[128 + tid] = w.c1b[tid];
      biasS[192 + tid] = w.c2w[tid]; biasS[256 + tid] = w.n1b[tid];
      biasS[320 + tid] = w.n2b[tid];
    }
    asyncWait0();
  }
  __syncthreads();

  const int wave = tid >> 5, lane = tid & 31, hi = lane >> 4, lrow = lane & 15;
  h16*   mA    = (h16*)(pws + (size_t)wave * (2048 + 256 + 64));
  float* xdT   = (float*)((char*)mA + 2048);
  float* coefT = xdT + 64;

  // ---- Phase 1: edge MLPs + segment sums (dst edges are contiguous) ----
  for (int ni = 0; ni < 16; ++ni) {
    const int n = n0 + wave * 16 + ni;
    const long eNode = (long)(g0 + n) * DEG;
    __builtin_prefetch(edge_src + eNode + DEG, 0, 0);
    float hagg[4] = {0.f, 0.f, 0.f, 0.f};
    float xagg0 = 0.f, xagg1 = 0.f, xagg2 = 0.f;

    for (int t = 0; t < 2; ++t) {
      // per-lane row data (lanes L and L+16 own the same edge row)
      const long e = eNode + t * 16 + lrow;
      const int srcl = edge_src[e] - g0;
      const float dx = xT[srcl * 4 + 0] - xT[n * 4 + 0];
      const float dy = xT[srcl * 4 + 1] - xT[n * 4 + 1];
      const float dz = xT[srcl * 4 + 2] - xT[n * 4 + 2];
      const float radial = dx * dx + dy * dy + dz * dz;
      const float inv = 1.f / (sqrtf(radial) + 1e-30f);
      if (hi) {
        xdT[lrow * 4 + 0] = dx * inv;
        xdT[lrow * 4 + 1] = dy * inv;
        xdT[lrow * 4 + 2] = dz * inv;
      }
      const float ef = edge_feat[e];

      // e1 A fragments gathered directly (no staging); bias slot carries 1.0
      v16h af[KC1];
      if constexpr (DIN == 64) {
        const h16* srow = hT + srcl * 64;
        const h16* drow = hT + n * 64;
        af[0] = rowAfrag(srow, 0, hi);
        af[1] = rowAfrag(srow, 32, hi);
        af[2] = rowAfrag(drow, 0, hi);
        af[3] = rowAfrag(drow, 32, hi);
        v16h a4 = zero16();
        if (!hi) { a4[0] = (h16)radial; a4[1] = (h16)ef; a4[2] = (h16)1.f; }
        af[4] = a4;
      } else {
        const h16* srow = hT + srcl * DIN;
        const h16* drow = hT + n * DIN;
        af[0] = gatherEdgeAfrag<DIN>(srow, drow, radial, ef, 0, hi);
        af[1] = gatherEdgeAfrag<DIN>(srow, drow, radial, ef, 32, hi);
      }
      // e1: [16,FK] x [FK,64] (+bias via folded row) -> silu -> mA
#pragma unroll
      for (int nt = 0; nt < 4; ++nt) {
        v8f c = zero8();
#pragma unroll
        for (int kc = 0; kc < KC1; ++kc)
          c = wmma16(af[kc], bfragSwz(e1Ws, kc, nt, lane), c);
        float tv[8];
#pragma unroll
        for (int r = 0; r < 8; ++r) tv[r] = siluf(c[r]);
        scatterColToA(mA, nt, lrow, hi, tv);
      }
      // e2: message m; per-column sums (h_agg) kept in registers
      v16h am0 = *(const v16h*)(mA + (size_t)lane * 16);
      v16h am1 = *(const v16h*)(mA + (size_t)(32 + lane) * 16);
      float msil[4][8];
#pragma unroll
      for (int nt = 0; nt < 4; ++nt) {
        v8f c = zero8();
        c = wmma16(am0, bfragSwz(e2Ws, 0, nt, lane), c);
        c = wmma16(am1, bfragSwz(e2Ws, 1, nt, lane), c);
        const float b = biasS[64 + nt * 16 + lrow];
        float s = 0.f;
#pragma unroll
        for (int r = 0; r < 8; ++r) { const float v = siluf(c[r] + b); msil[nt][r] = v; s += v; }
        hagg[nt] += s;
      }
#pragma unroll
      for (int nt = 0; nt < 4; ++nt) scatterColToA(mA, nt, lrow, hi, msil[nt]);
      // c1 + c2 (dot with c2w over the 64 columns)
      v16h ac0 = *(const v16h*)(mA + (size_t)lane * 16);
      v16h ac1 = *(const v16h*)(mA + (size_t)(32 + lane) * 16);
      float cp[8] = {0.f,0.f,0.f,0.f,0.f,0.f,0.f,0.f};
#pragma unroll
      for (int nt = 0; nt < 4; ++nt) {
        v8f c = zero8();
        c = wmma16(ac0, bfragSwz(c1Ws, 0, nt, lane), c);
        c = wmma16(ac1, bfragSwz(c1Ws, 1, nt, lane), c);
        const float b  = biasS[128 + nt * 16 + lrow];
        const float w2 = biasS[192 + nt * 16 + lrow];
#pragma unroll
        for (int r = 0; r < 8; ++r) cp[r] += siluf(c[r] + b) * w2;
      }
#pragma unroll
      for (int off = 1; off < 16; off <<= 1)
#pragma unroll
        for (int r = 0; r < 8; ++r) cp[r] += __shfl_xor(cp[r], off, 32);
      if (lrow == 0)
#pragma unroll
        for (int r = 0; r < 8; ++r) coefT[r + 8 * hi] = cp[r];
      { // x_agg += sum over 16 edges of coef * xd
        float px = 0.f, py = 0.f, pz = 0.f;
        if (!hi) {
          const float cf = coefT[lrow];
          px = cf * xdT[lrow * 4 + 0];
          py = cf * xdT[lrow * 4 + 1];
          pz = cf * xdT[lrow * 4 + 2];
        }
#pragma unroll
        for (int off = 1; off < 32; off <<= 1) {
          px += __shfl_xor(px, off, 32);
          py += __shfl_xor(py, off, 32);
          pz += __shfl_xor(pz, off, 32);
        }
        xagg0 += px; xagg1 += py; xagg2 += pz;
      }
    } // edge sub-tiles
#pragma unroll
    for (int nt = 0; nt < 4; ++nt) hagg[nt] += __shfl_xor(hagg[nt], 16, 32);
    if (!hi)
#pragma unroll
      for (int nt = 0; nt < 4; ++nt)
        haggT[(n - n0) * 64 + nt * 16 + lrow] = (h16)hagg[nt];
    if (lane == 0) {
      const size_t g = (size_t)(g0 + n);
      x_out[g * 4 + 0] = xT[n * 4 + 0] + xagg0;
      x_out[g * 4 + 1] = xT[n * 4 + 1] + xagg1;
      x_out[g * 4 + 2] = xT[n * 4 + 2] + xagg2;
      x_out[g * 4 + 3] = 0.f;
    }
  }
  __syncthreads();

  // ---- Phase 2: node MLP (one 16-node tile per wave) ----
  {
    const int nb = n0 + wave * 16;
    const h16* hrow = hT + (size_t)(nb + lrow) * DIN;
    const h16* grow = haggT + (size_t)(nb + lrow - n0) * 64;
    v16h af[KC2];
    if constexpr (DIN == 64) {
      af[0] = rowAfrag(hrow, 0, hi);
      af[1] = rowAfrag(hrow, 32, hi);
      af[2] = rowAfrag(grow, 0, hi);
      af[3] = rowAfrag(grow, 32, hi);
    } else {
      af[0] = gatherNodeAfrag<DIN>(hrow, grow, 0, hi);
      af[1] = gatherNodeAfrag<DIN>(hrow, grow, 32, hi);
      af[2] = gatherNodeAfrag<DIN>(hrow, grow, 64, hi);
    }
#pragma unroll
    for (int nt = 0; nt < 4; ++nt) {
      v8f c = zero8();
#pragma unroll
      for (int kc = 0; kc < KC2; ++kc)
        c = wmma16(af[kc], bfragSwz(n1Ws, kc, nt, lane), c);
      const float b = biasS[256 + nt * 16 + lrow];
      float tv[8];
#pragma unroll
      for (int r = 0; r < 8; ++r) tv[r] = siluf(c[r] + b);
      scatterColToA(mA, nt, lrow, hi, tv);
    }
    v16h a0 = *(const v16h*)(mA + (size_t)lane * 16);
    v16h a1 = *(const v16h*)(mA + (size_t)(32 + lane) * 16);
#pragma unroll
    for (int nt = 0; nt < 4; ++nt) {
      v8f c = zero8();
      c = wmma16(a0, bfragSwz(n2Ws, 0, nt, lane), c);
      c = wmma16(a1, bfragSwz(n2Ws, 1, nt, lane), c);
      const float b = biasS[320 + nt * 16 + lrow];
#pragma unroll
      for (int r = 0; r < 8; ++r) {
        const int node = nb + r + 8 * hi;
        h_out[((size_t)g0 + node) * 64 + nt * 16 + lrow] = (h16)(c[r] + b);
      }
    }
  }
}

// --------------------------- MHA + mean pool -------------------------------

constexpr size_t mha_lds_bytes() {
  return (size_t)512 * 64 * 2 * 2      // hT, kT
       + (size_t)64 * 512 * 2          // vTt (head-transposed V)
       + (size_t)128 * 64 * 2          // qT (this part's queries, pre-scaled)
       + (size_t)3 * 4096 * 2          // swizzled q/k/v weights
       + (size_t)192 * 4 + 64 * 4      // biases + pool
       + (size_t)8 * 512 * 2;          // per-wave P staging (A-ready)
}

// Q A-fragment: only lanes 0-15, elements 0-7 (d=0..7) nonzero -> one b128
__device__ __forceinline__ v16h qfragAv(const h16* qT, int row0, int h8, int lane) {
  v16h a = zero16();
  if (lane < 16) {
    union { uint4 q; h16 e[8]; } u;
    u.q = *(const uint4*)(qT + (size_t)(row0 + lane) * 64 + h8);
#pragma unroll
    for (int i = 0; i < 8; ++i) a[i] = u.e[i];
  }
  return a;
}
// K^T B-fragment: B[d][key]; only lanes 0-15, elements 0-7 (d=0..7) carry data -> one b128
__device__ __forceinline__ v16h kfragBv(const h16* kT, int key0, int h8, int lane) {
  v16h b = zero16();
  if (lane < 16) {
    union { uint4 q; h16 e[8]; } u;
    u.q = *(const uint4*)(kT + (size_t)(key0 + lane) * 64 + h8);
#pragma unroll
    for (int i = 0; i < 8; ++i) b[i] = u.e[i];
  }
  return b;
}
// V B-fragment from head-transposed vTt[(h8+d)*512 + key]: contiguous in key
__device__ __forceinline__ v16h vfragTv(const h16* vTt, int key0, int h8, int lane) {
  const int col = lane & 15;
  const int kb  = (lane >> 4) * 8;
  if (col < 8) {
    union { v16h v; uint4 q[2]; } u;
    const h16* p = vTt + (size_t)(h8 + col) * 512 + key0 + kb;
    u.q[0] = *(const uint4*)(p);
    u.q[1] = *(const uint4*)(p + 16);
    return u.v;
  }
  return zero16();
}

__global__ __launch_bounds__(256) void mha_pool_kernel(
    const h16* __restrict__ h_in,
    const h16* __restrict__ qW, const h16* __restrict__ kW, const h16* __restrict__ vW,
    const float* __restrict__ qb, const float* __restrict__ kb2, const float* __restrict__ vb,
    float* __restrict__ poolO)
{
  extern __shared__ char smem[];
  h16*   hT    = (h16*)smem;            // 512x64
  h16*   kT    = hT + 512 * 64;         // 512x64
  h16*   vTt   = kT + 512 * 64;         // 64 x 512 (head-dim major)
  h16*   qT    = vTt + 64 * 512;        // 128x64 (this part, pre-scaled)
  h16*   wS    = qT + 128 * 64;         // 3 x swizzled 64x64
  float* bS    = (float*)(wS + 3 * 4096);
  float* poolL = bS + 192;              // 64
  h16*   pA0   = (h16*)(poolL + 64);    // 8 waves x [32 lanes][16]

  const int tid   = threadIdx.x;
  const int graph = blockIdx.x >> 2;
  const int part  = blockIdx.x & 3;
  const int g0    = graph * NGRAPH;
  const int q0    = part * 128;

  { // async DMA staging
    const uint4* hs = (const uint4*)(h_in + (size_t)g0 * 64);
    uint4* hd = (uint4*)hT;
    for (int i = tid; i < 512 * 64 / 8; i += 256) asyncLds128(hd + i, hs + i);
    const uint4* q4 = (const uint4*)qW; const uint4* k4 = (const uint4*)kW;
    const uint4* v4 = (const uint4*)vW; uint4* wd = (uint4*)wS;
    for (int i = tid; i < 512; i += 256) {
      asyncLds128(wd + i, q4 + i);
      asyncLds128(wd + 512 + i, k4 + i);
      asyncLds128(wd + 1024 + i, v4 + i);
    }
    if (tid < 64) { bS[tid] = qb[tid]; bS[64 + tid] = kb2[tid]; bS[128 + tid] = vb[tid]; }
    if (tid < 64) poolL[tid] = 0.f;
    asyncWait0();
  }
  __syncthreads();

  const int wave = tid >> 5, lane = tid & 31, hi = lane >> 4, lrow = lane & 15;

  // Q/K/V projections (K,V full graph; Q only this part's rows)
  for (int tile = wave * 4; tile < wave * 4 + 4; ++tile) {
    const int nb = tile * 16;
    const h16* hrow = hT + (size_t)(nb + lrow) * 64;
    v16h a0 = rowAfrag(hrow, 0, hi);
    v16h a1 = rowAfrag(hrow, 32, hi);
    for (int m = 0; m < 3; ++m) {
      if (m == 0 && (nb < q0 || nb >= q0 + 128)) continue;
      const h16* W = wS + m * 4096;
#pragma unroll
      for (int nt = 0; nt < 4; ++nt) {
        v8f c = zero8();
        c = wmma16(a0, bfragSwz(W, 0, nt, lane), c);
        c = wmma16(a1, bfragSwz(W, 1, nt, lane), c);
        const float b = bS[m * 64 + nt * 16 + lrow];
        const int col = nt * 16 + lrow;
        if (m == 2) {           // V: packed store, head-transposed layout
          union { uint4 q; h16 e[8]; } u;
#pragma unroll
          for (int r = 0; r < 8; ++r) u.e[r] = (h16)(c[r] + b);
          *(uint4*)(vTt + (size_t)col * 512 + nb + 8 * hi) = u.q;
        } else {
#pragma unroll
          for (int r = 0; r < 8; ++r) {
            const int node = nb + r + 8 * hi;
            const float val = c[r] + b;
            if (m == 0) qT[(size_t)(node - q0) * 64 + col] = (h16)(val * 0.35355339f);
            else        kT[(size_t)node * 64 + col] = (h16)val;
          }
        }
      }
    }
  }
  __syncthreads();

  // Flash-style attention: (head, 16-query tile) items, 512 keys each
  h16* pA = pA0 + wave * 512;
  const int hiMp = (lrow >> 3) & 1;        // P-scatter constants for this lane
  const int e0p  = lrow - 8 * hiMp;
  for (int item = wave; item < 64; item += 8) {
    const int head = item >> 3, qt = item & 7;
    const int h8 = head * 8;
    v16h aq = qfragAv(qT, qt * 16, h8, lane);
    float m_[8], l_[8];
#pragma unroll
    for (int r = 0; r < 8; ++r) { m_[r] = -1e30f; l_[r] = 0.f; }
    v8f Oacc = zero8();
    for (int kc = 0; kc < 512; kc += 32) {
      v8f s0 = zero8(), s1 = zero8();
      s0 = wmma16(aq, kfragBv(kT, kc,      h8, lane), s0);
      s1 = wmma16(aq, kfragBv(kT, kc + 16, h8, lane), s1);
      float cm[8];
#pragma unroll
      for (int r = 0; r < 8; ++r) cm[r] = fmaxf(s0[r], s1[r]);
#pragma unroll
      for (int off = 1; off < 16; off <<= 1)
#pragma unroll
        for (int r = 0; r < 8; ++r) cm[r] = fmaxf(cm[r], __shfl_xor(cm[r], off, 32));
      float rs[8];
#pragma unroll
      for (int r = 0; r < 8; ++r) {
        const float nm = fmaxf(m_[r], cm[r]);
        const float sc = __expf(m_[r] - nm);
        const float p0 = __expf(s0[r] - nm);
        const float p1 = __expf(s1[r] - nm);
        const int laneM = (r + 8 * hi) + 16 * hiMp;
        pA[laneM * 16 + e0p]     = (h16)p0;
        pA[laneM * 16 + e0p + 8] = (h16)p1;
        rs[r] = p0 + p1;
        Oacc[r] *= sc;
        l_[r] = l_[r] * sc;
        m_[r] = nm;
      }
#pragma unroll
      for (int off = 1; off < 16; off <<= 1)
#pragma unroll
        for (int r = 0; r < 8; ++r) rs[r] += __shfl_xor(rs[r], off, 32);
#pragma unroll
      for (int r = 0; r < 8; ++r) l_[r] += rs[r];
      v16h ap = *(const v16h*)(pA + (size_t)lane * 16);
      Oacc = wmma16(ap, vfragTv(vTt, kc, h8, lane), Oacc);
    }
    float s = 0.f;
#pragma unroll
    for (int r = 0; r < 8; ++r) s += Oacc[r] / l_[r];
    s += __shfl_xor(s, 16, 32);
    if (lane < 8) atomicAdd(&poolL[h8 + lane], s);
  }
  __syncthreads();
  if (tid < 64) atomicAdd(&poolO[graph * 64 + tid], poolL[tid] * (1.f / 512.f));
}

// ------------------------ pooled heads (tiny) ------------------------------

__global__ __launch_bounds__(64) void head_kernel(
    const float* __restrict__ poolO,
    const float* oW, const float* ob,
    const float* clsW, const float* clsb,
    const float* headW, const float* headb,
    const float* nodeW, const float* nodeb,
    float* __restrict__ out)
{
  const int b = threadIdx.x;
  if (b >= 64) return;
  float att[64];
  for (int j = 0; j < 64; ++j) {
    float s = ob[j];
    for (int i = 0; i < 64; ++i) s += poolO[b * 64 + i] * oW[i * 64 + j];
    att[j] = s;
  }
  float feat[32];
  for (int j = 0; j < 32; ++j) {
    float s = clsb[j];
    for (int i = 0; i < 64; ++i) s += att[i] * clsW[i * 32 + j];
    feat[j] = fmaxf(s, 0.f);
  }
  float fo = headb[0];
  for (int i = 0; i < 32; ++i) fo += feat[i] * headW[i];
  out[b] = fo;
  for (int j = 0; j < 20; ++j) {
    float s = nodeb[j];
    for (int i = 0; i < 32; ++i) s += feat[i] * nodeW[i * 20 + j];
    out[64 + b * 20 + j] = s;
  }
}

// ------------------------------ launch -------------------------------------

extern "C" void kernel_launch(void* const* d_in, const int* in_sizes, int n_in,
                              void* d_out, int out_size, void* d_ws, size_t ws_size,
                              hipStream_t stream)
{
  (void)in_sizes; (void)n_in; (void)out_size; (void)ws_size;
  // pytree leaf order: 0..6 top-level arrays, then params:
  //   attn (k.W,k.b,o.W,o.b,q.W,q.b,v.W,v.b) = 7..14
  //   egnn[l] base 15+11l: c1.W,c1.b,c2.W,e1.W,e1.b,e2.W,e2.b,n1.W,n1.b,n2.W,n2.b
  //   mlp: 81 cls.W, 82 cls.b, 83 head.W, 84 head.b, 85 node.W, 86 node.b
  const float* node_feat = (const float*)d_in[0];
  const float* coord     = (const float*)d_in[1];
  const float* edge_feat = (const float*)d_in[2];
  const int*   edge_src  = (const int*)d_in[5];
  auto P = [&](int i) { return (const float*)d_in[i]; };

  char* ws = (char*)d_ws;
  size_t off = 0;
  auto alloc = [&](size_t bytes) {
    char* p = ws + off; off += (bytes + 255) & ~(size_t)255; return p;
  };
  h16*   HB0 = (h16*)alloc((size_t)NODES_TOT * 64 * 2);
  h16*   HB1 = (h16*)alloc((size_t)NODES_TOT * 64 * 2);
  float* XB0 = (float*)alloc((size_t)NODES_TOT * 4 * 4);
  float* XB1 = (float*)alloc((size_t)NODES_TOT * 4 * 4);
  h16 *E1[6], *E2[6], *C1[6], *N1[6], *N2[6];
  for (int l = 0; l < 6; ++l) {
    const int FK = l ? 160 : 64, KN1 = l ? 128 : 96;
    E1[l] = (h16*)alloc((size_t)FK * 64 * 2);
    E2[l] = (h16*)alloc(4096 * 2);
    C1[l] = (h16*)alloc(4096 * 2);
    N1[l] = (h16*)alloc((size_t)KN1 * 64 * 2);
    N2[l] = (h16*)alloc(4096 * 2);
  }
  h16* QW = (h16*)alloc(8192);
  h16* KW = (h16*)alloc(8192);
  h16* VW = (h16*)alloc(8192);
  float* POOL = (float*)alloc(64 * 64 * 4);

  PrepArgs pa{};
  pa.node_feat = node_feat; pa.coord = coord;
  for (int l = 0; l < 6; ++l) {
    const int base = 15 + 11 * l;
    pa.c1W[l] = P(base + 0); pa.e1W[l] = P(base + 3); pa.e2W[l] = P(base + 5);
    pa.n1W[l] = P(base + 7); pa.n2W[l] = P(base + 9);
    pa.e1b[l] = P(base + 4);
    pa.e1o[l] = E1[l]; pa.e2o[l] = E2[l]; pa.c1o[l] = C1[l];
    pa.n1o[l] = N1[l]; pa.n2o[l] = N2[l];
  }
  pa.qW = P(11); pa.kW = P(7); pa.vW = P(13);
  pa.h0 = HB0; pa.x0 = XB0;
  pa.qo = QW; pa.ko = KW; pa.vo = VW; pa.poolO = POOL;
  prep_kernel<<<512, 256, 0, stream>>>(pa);

  const size_t lds0 = edge_lds_bytes(20, 64, 96);
  const size_t ldsN = edge_lds_bytes(64, 160, 128);
  const size_t ldsM = mha_lds_bytes();
  hipFuncSetAttribute((const void*)egnn_layer_kernel<20, 64, 96>,
                      hipFuncAttributeMaxDynamicSharedMemorySize, (int)lds0);
  hipFuncSetAttribute((const void*)egnn_layer_kernel<64, 160, 128>,
                      hipFuncAttributeMaxDynamicSharedMemorySize, (int)ldsN);
  hipFuncSetAttribute((const void*)mha_pool_kernel,
                      hipFuncAttributeMaxDynamicSharedMemorySize, (int)ldsM);

  auto mkLW = [&](int l) -> LayerW {
    const int base = 15 + 11 * l;
    LayerW w;
    w.e1W = E1[l]; w.e2W = E2[l]; w.c1W = C1[l]; w.n1W = N1[l]; w.n2W = N2[l];
    w.e2b = P(base + 6); w.c1b = P(base + 1);
    w.c2w = P(base + 2); w.n1b = P(base + 8); w.n2b = P(base + 10);
    return w;
  };

  egnn_layer_kernel<20, 64, 96><<<BGRAPH * NSPLIT, 256, lds0, stream>>>(
      HB0, XB0, HB1, XB1, edge_src, edge_feat, mkLW(0));
  h16* hin = HB1; float* xin = XB1; h16* hout = HB0; float* xout = XB0;
  for (int l = 1; l < 6; ++l) {
    egnn_layer_kernel<64, 160, 128><<<BGRAPH * NSPLIT, 256, ldsN, stream>>>(
        hin, xin, hout, xout, edge_src, edge_feat, mkLW(l));
    h16* th = hin; hin = hout; hout = th;
    float* tx = xin; xin = xout; xout = tx;
  }
  // after the loop, `hin` points at the final h (HB0)
  mha_pool_kernel<<<BGRAPH * NSPLIT, 256, ldsM, stream>>>(
      hin, QW, KW, VW, P(12), P(8), P(14), POOL);
  head_kernel<<<1, 64, 0, stream>>>(
      POOL, P(9), P(10), P(81), P(82), P(83), P(84), P(85), P(86), (float*)d_out);
}
